// SSDAttention_78864189489521
// MI455X (gfx1250) — matrix operand order repaired and checked by reading, not compiled, not run on hardware
//
#include <hip/hip_runtime.h>

typedef __attribute__((ext_vector_type(16))) __bf16 v16bf;
typedef __attribute__((ext_vector_type(8)))  float  v8f;

#define BATCH   4
#define SEQ     4096
#define DMODEL  1024
#define NHEADS  16
#define HEADDIM 64
#define NCHUNK  64
#define BLK     64

__device__ __forceinline__ __bf16 f2bf(float f) {
  union { float f; unsigned u; } x; x.f = f;
  unsigned r = (x.u + 0x7FFFu + ((x.u >> 16) & 1u)) >> 16;
  unsigned short s = (unsigned short)r;
  __bf16 b;
  __builtin_memcpy(&b, &s, 2);
  return b;
}

// ---------------- Stage 0: proj_w (f32) -> bf16 ----------------
__global__ void convert_w_kernel(const float* __restrict__ W,
                                 __bf16* __restrict__ Wbf, int n) {
  int i = blockIdx.x * blockDim.x + threadIdx.x;
  if (i < n) Wbf[i] = f2bf(W[i]);
}

// ---------------- Stage 1: chunk-state scan (parallelized Horner) ----------
// S[b,c,h,p]  = sum_l exp(A_h*(63-l)) x[b,c,l,h,p]
//             = sum_{j=0..3} f^{16(3-j)} * Horner_{i<16}(x[16j+i])
// ns[b,c,h,p]: ns[0]=ns[1]=0 ; ns[c+1] = e^{64A}(ns[c] + S[c-1])
// One block per (b,h); 4 l-groups x 64 p-lanes; recurrence held by group 0.
__global__ void __launch_bounds__(256)
chunk_state_kernel(const float* __restrict__ x,
                   const float* __restrict__ A_w,
                   float* __restrict__ ns_out) {
  const int h = blockIdx.x & 15;
  const int b = blockIdx.x >> 4;
  const int tid = threadIdx.x;
  const int p = tid & 63;
  const int j = tid >> 6;                       // l-group 0..3
  const float A   = A_w[h];
  const float f   = __expf(A);
  const float e64 = __expf(64.f * A);
  const float fg  = __expf(A * (float)(16 * (3 - j)));  // f^{16(3-j)}

  __shared__ float part[4][64];

  float ns = 0.f, Sm1 = 0.f;
  const float* xb = x + (size_t)b * SEQ * DMODEL + h * HEADDIM + p;
  for (int c = 0; c < NCHUNK; ++c) {
    const float* xc = xb + (size_t)(c * BLK + j * 16) * DMODEL;
    float P = 0.f;
#pragma unroll
    for (int i = 0; i < 16; ++i) P = P * f + xc[(size_t)i * DMODEL];
    part[j][p] = P * fg;
    __syncthreads();
    if (j == 0) {
      float S = part[0][p] + part[1][p] + part[2][p] + part[3][p];
      ns_out[((b * NCHUNK + c) * NHEADS + h) * HEADDIM + p] = ns;
      ns  = e64 * (ns + Sm1);
      Sm1 = S;
    }
    __syncthreads();
  }
}

// ---------------- Stage 2: intra (L @ X) + inter fuse, emit bf16 y ----------------
// One block per (b,c,h). LDS: L (bf16, 64x64) row-major, X^T (bf16, p-major).
// 8 waves x 2 tiles -> 16 tiles of 16x16 via v_wmma_f32_16x16x32_bf16.
__global__ void __launch_bounds__(256)
intra_kernel(const float* __restrict__ x, const float* __restrict__ A_w,
             const float* __restrict__ B_w, const float* __restrict__ C_w,
             const float* __restrict__ nsbuf, __bf16* __restrict__ ybf) {
  const int h = blockIdx.x & 15;
  const int c = (blockIdx.x >> 4) & 63;
  const int b = blockIdx.x >> 10;

  __shared__ __align__(16) __bf16 Lbf[64 * 72];
  __shared__ __align__(16) __bf16 XT [64 * 72];
  __shared__ float nsS[64];

  const int tid = threadIdx.x;
  const float A = A_w[h];
  float wh = 0.f;
#pragma unroll 8
  for (int n = 0; n < 64; ++n) wh += B_w[h * 64 + n] * C_w[h * 64 + n];

  const float* xsrc = x + ((size_t)(b * NCHUNK + c) * BLK) * DMODEL + h * HEADDIM;
  for (int i = tid; i < 4096; i += 256) {
    int l = i >> 6, pp = i & 63;
    float xv = xsrc[(size_t)l * DMODEL + pp];
    XT[pp * 72 + l] = f2bf(xv);                      // store transposed: [p][s]
    float lv = (pp < l) ? __expf(A * (float)(l - pp)) : 0.f;  // strict lower tri
    Lbf[l * 72 + pp] = f2bf(lv);
  }
  if (tid < 64)
    nsS[tid] = nsbuf[((b * NCHUNK + c) * NHEADS + h) * HEADDIM + tid];
  __syncthreads();

  const int lane = tid & 31;
  const int wave = tid >> 5;
  union U { v16bf v; uint4 q[2]; };

  for (int qq = 0; qq < 2; ++qq) {
    int t = wave * 2 + qq;           // 0..15
    int it = t >> 2, jt = t & 3;     // 4x4 tiles of 16x16
    v8f acc = {};
#pragma unroll
    for (int kt = 0; kt < 2; ++kt) {
      int k0 = kt * 32;
      U a, bm;
      // A fragment: row m = it*16 + lane%16; halves 0-7: K=k0+kg..+7, 8-15: K=k0+16+kg..+7
      int m  = it * 16 + (lane & 15);
      int kg = (lane >> 4) * 8;
      a.q[0] = *(const uint4*)&Lbf[m * 72 + k0 + kg];
      a.q[1] = *(const uint4*)&Lbf[m * 72 + k0 + 16 + kg];
      // B fragment: col n = jt*16 + lane%16; halves 0-7: K=k0+kg..+7, 8-15: K=k0+16+kg..+7
      int n   = jt * 16 + (lane & 15);
      int kgb = (lane >> 4) * 8;
      bm.q[0] = *(const uint4*)&XT[n * 72 + k0 + kgb];
      bm.q[1] = *(const uint4*)&XT[n * 72 + k0 + 16 + kgb];
      acc = __builtin_amdgcn_wmma_f32_16x16x32_bf16(false, a.v, false, bm.v,
                                                    (short)0, acc, false, false);
    }
    // epilogue: y = w_h * (intra + e^{A(l+1)} * ns[p])
    int pcol = jt * 16 + (lane & 15);
    float nsv = nsS[pcol];
#pragma unroll
    for (int r = 0; r < 8; ++r) {
      int l = it * 16 + r + 8 * (lane >> 4);
      float e = __expf(A * (float)(l + 1));
      float yv = wh * (acc[r] + e * nsv);
      ybf[((size_t)(b * NCHUNK + c) * BLK + l) * DMODEL + h * HEADDIM + pcol] = f2bf(yv);
    }
  }
}

// ---------------- Stage 3: out = Y(bf16) @ W(bf16)^T + bias, f32 out ----------------
// Register-blocked: one wave per 32(M) x 64(N) region = 2x4 tiles of 16x16.
// Per k-step: 2 A-fragments + 4 B-fragments -> 8 v_wmma_f32_16x16x32_bf16.
__global__ void __launch_bounds__(256)
gemm_kernel(const __bf16* __restrict__ Y, const __bf16* __restrict__ W,
            const float* __restrict__ bias, float* __restrict__ out) {
  const int lane = threadIdx.x & 31;
  const int wave = threadIdx.x >> 5;
  const int wg   = blockIdx.x * 8 + wave;       // global wave id
  const int SN   = DMODEL / 64;                 // 16 N-supertiles
  const int sm   = wg / SN;                     // 0..511 (M supertile, 32 rows)
  const int sn   = wg % SN;                     // 0..15  (N supertile, 64 cols)

  const int kgA = (lane >> 4) * 8;
  const int kgB = (lane >> 4) * 8;
  const __bf16* arow0 = Y + (size_t)(sm * 32 + (lane & 15)) * DMODEL;
  const __bf16* arow1 = arow0 + (size_t)16 * DMODEL;
  const __bf16* brow0 = W + (size_t)(sn * 64 + (lane & 15)) * DMODEL;

  v8f acc[8] = {};
  union U { v16bf v; uint4 q[2]; };
#pragma unroll 2
  for (int k0 = 0; k0 < DMODEL; k0 += 32) {
    __builtin_prefetch(arow0 + k0 + 512, 0, 1);   // global_prefetch_b8
    __builtin_prefetch(arow1 + k0 + 512, 0, 1);
    U a0, a1, bm[4];
    a0.q[0] = *(const uint4*)&arow0[k0 + kgA];
    a0.q[1] = *(const uint4*)&arow0[k0 + 16 + kgA];
    a1.q[0] = *(const uint4*)&arow1[k0 + kgA];
    a1.q[1] = *(const uint4*)&arow1[k0 + 16 + kgA];
#pragma unroll
    for (int j = 0; j < 4; ++j) {
      const __bf16* br = brow0 + (size_t)(j * 16) * DMODEL;
      bm[j].q[0] = *(const uint4*)&br[k0 + kgB];
      bm[j].q[1] = *(const uint4*)&br[k0 + 16 + kgB];
    }
#pragma unroll
    for (int j = 0; j < 4; ++j) {
      acc[j]     = __builtin_amdgcn_wmma_f32_16x16x32_bf16(false, a0.v, false, bm[j].v,
                                                           (short)0, acc[j], false, false);
      acc[4 + j] = __builtin_amdgcn_wmma_f32_16x16x32_bf16(false, a1.v, false, bm[j].v,
                                                           (short)0, acc[4 + j], false, false);
    }
  }
#pragma unroll
  for (int i = 0; i < 2; ++i) {
#pragma unroll
    for (int j = 0; j < 4; ++j) {
      const int n = sn * 64 + j * 16 + (lane & 15);
      const float bv = bias[n];
#pragma unroll
      for (int r = 0; r < 8; ++r) {
        int mr = sm * 32 + i * 16 + r + 8 * (lane >> 4);
        out[(size_t)mr * DMODEL + n] = acc[i * 4 + j][r] + bv;
      }
    }
  }
}

extern "C" void kernel_launch(void* const* d_in, const int* in_sizes, int n_in,
                              void* d_out, int out_size, void* d_ws, size_t ws_size,
                              hipStream_t stream) {
  const float* x      = (const float*)d_in[0];
  const float* A_w    = (const float*)d_in[1];
  const float* B_w    = (const float*)d_in[2];
  const float* C_w    = (const float*)d_in[3];
  const float* proj_w = (const float*)d_in[4];
  const float* proj_b = (const float*)d_in[5];
  float* out = (float*)d_out;

  char* ws = (char*)d_ws;
  __bf16* ybf   = (__bf16*)ws;                                 // 32 MB
  __bf16* Wbf   = (__bf16*)(ws + (size_t)32 * 1024 * 1024);    // 2 MB
  float*  nsbuf = (float*)(ws + (size_t)34 * 1024 * 1024);     // 1 MB

  convert_w_kernel<<<4096, 256, 0, stream>>>(proj_w, Wbf, DMODEL * DMODEL);
  chunk_state_kernel<<<BATCH * NHEADS, 256, 0, stream>>>(x, A_w, nsbuf);
  intra_kernel<<<BATCH * NCHUNK * NHEADS, 256, 0, stream>>>(x, A_w, B_w, C_w, nsbuf, ybf);
  // 512 M-supertiles (32 rows) x 16 N-supertiles (64 cols) = 8192 waves / 8 = 1024 blocks
  gemm_kernel<<<1024, 256, 0, stream>>>(ybf, Wbf, proj_b, out);
}
